// SceneGraphModel_47665547051599
// MI455X (gfx1250) — matrix operand order repaired and checked by reading, not compiled, not run on hardware
//
#include <hip/hip_runtime.h>
#include <stdint.h>

typedef __attribute__((ext_vector_type(16))) __bf16 bf16x16;
typedef __attribute__((ext_vector_type(8)))  __bf16 bf16x8;
typedef __attribute__((ext_vector_type(8)))  float  v8f;

// ---------------- workspace layout (element offsets in __bf16) ----------------
constexpr int OFF_W1  = 0;          // 512x256 frag-ordered
constexpr int OFF_W2  = 131072;     // 256x128
constexpr int OFF_W3  = 163840;     // 128x64
constexpr int OFF_W4  = 172032;     // 64x16
constexpr int OFF_W5  = 173056;     // 32x16 (zero-padded from 16x8)
constexpr int OFF_WFC = 173568;     // 128x64
constexpr int WS_BF16_TOTAL = 181760;
constexpr size_t OFF_EDGE_BYTES = (size_t)WS_BF16_TOTAL * 2;   // one float

// -------- CDNA5 async global->LDS path (guarded; falls back to sync copy) -----
#if defined(__gfx1250__) && __has_builtin(__builtin_amdgcn_global_load_async_to_lds_b128)
#define USE_ASYNC 1
#else
#define USE_ASYNC 0
#endif

#if USE_ASYNC
#if __has_builtin(__builtin_amdgcn_s_wait_asynccnt)
#define WAIT_ASYNC(n) __builtin_amdgcn_s_wait_asynccnt(n)
#else
#define WAIT_ASYNC(n) asm volatile("s_wait_asynccnt %0" :: "i"(n) : "memory")
#endif
typedef __attribute__((ext_vector_type(4))) int i32x4;
typedef __attribute__((address_space(3))) i32x4 lds_i32x4;
// stage one 16KB chunk: 256 threads x 4 async b128 (64B/thread)
__device__ __forceinline__ void stage_issue(const __bf16* g, __bf16* l, int tid) {
  #pragma unroll
  for (int i = 0; i < 4; ++i) {
    __builtin_amdgcn_global_load_async_to_lds_b128(
        (i32x4*)(g + tid * 32 + i * 8),
        (lds_i32x4*)(uintptr_t)(l + tid * 32 + i * 8),
        0, 0);
  }
}
#else
__device__ __forceinline__ void stage_copy(const __bf16* g, __bf16* l, int tid) {
  const bf16x16* gs = (const bf16x16*)g + (size_t)tid * 2;
  bf16x16* ls = (bf16x16*)l + (size_t)tid * 2;
  const bf16x16 t0 = gs[0];
  const bf16x16 t1 = gs[1];
  ls[0] = t0;
  ls[1] = t1;
}
#endif

// ---------------- prep: swizzle f32 weights into WMMA B-fragment order --------
// B-frag layout (bf16 32xK -> 16 cols): lane = column (lane&15), lanes 0-15 hold
// K=kt*32+0..15, lanes 16-31 hold K=kt*32+16..31, 16 contiguous bf16 per lane.
__global__ void prep_wfrag(const float* __restrict__ W, __bf16* __restrict__ dst,
                           int din, int dout, int kpad, int dpad) {
  const int ntiles = dpad >> 4;
  const int total  = kpad * dpad;
  for (int tid = blockIdx.x * blockDim.x + threadIdx.x; tid < total;
       tid += gridDim.x * blockDim.x) {
    const int tileIdx = tid >> 9;        // 512 elements per (kt,nt) tile
    const int within  = tid & 511;
    const int lane    = within >> 4;
    const int j       = within & 15;
    const int kt  = tileIdx / ntiles;
    const int nt  = tileIdx % ntiles;
    const int col = nt * 16 + (lane & 15);
    const int k   = kt * 32 + (lane >> 4) * 16 + j;
    float v = 0.f;
    if (k < din && col < dout) v = W[k * dout + col];
    dst[tid] = (__bf16)v;
  }
}

__global__ void prep_edge(const float* __restrict__ We, const float* __restrict__ ae,
                          float* __restrict__ out) {
  __shared__ float red[256];
  red[threadIdx.x] = We[threadIdx.x] * ae[threadIdx.x];
  __syncthreads();
  for (int o = 128; o > 0; o >>= 1) {
    if ((int)threadIdx.x < o) red[threadIdx.x] += red[threadIdx.x + o];
    __syncthreads();
  }
  if (threadIdx.x == 0) *out = red[0];
}

// ---------------- fused GAT layer (one wave == 2 batches == 16 rows) ----------
__device__ __forceinline__ float red16(float v) {
  v += __shfl_xor(v, 1, 32);
  v += __shfl_xor(v, 2, 32);
  v += __shfl_xor(v, 4, 32);
  v += __shfl_xor(v, 8, 32);
  return v;
}

template<int DIN, int DOUT, int VD, int OSTR, bool FIRST, bool ZPAD, bool LAST>
__device__ __forceinline__ void gat_layer(
    const __bf16* __restrict__ xb, __bf16* __restrict__ xout,
    const __bf16* __restrict__ wfrag,
    __bf16* __restrict__ stage, int tid,
    const float* __restrict__ a_s, const float* __restrict__ a_t,
    const float* __restrict__ bias,
    const float* __restrict__ roi, const float* __restrict__ prop,
    float edgec, int bbase, int lane, int lb0)
{
  constexpr int NT = DOUT / 16;
  constexpr int KC = DIN / 32;
  constexpr int CH = NT * 512;        // staged-chunk elements (FIRST layer only)
  const int lr = lane & 15;
  const int lh = lane >> 4;

  v8f acc[NT];
  #pragma unroll
  for (int t = 0; t < NT; ++t) acc[t] = {};

#if USE_ASYNC
  if constexpr (FIRST) stage_issue(wfrag, stage, tid);   // prologue: chunk 0
#endif

  for (int kc = 0; kc < KC; ++kc) {
    // --- block-cooperative staging of B fragments for the big first layer ---
    if constexpr (FIRST) {
#if USE_ASYNC
      if (kc + 1 < KC) {
        stage_issue(wfrag + (size_t)(kc + 1) * CH, stage + ((kc + 1) & 1) * CH, tid);
        WAIT_ASYNC(4);                  // chunk kc landed (4 newer still in flight)
      } else {
        WAIT_ASYNC(0);
      }
      __syncthreads();                  // whole chunk visible to all waves
#else
      __syncthreads();                  // previous readers done
      stage_copy(wfrag + (size_t)kc * CH, stage, tid);
      __syncthreads();
#endif
    }

    // A fragment: row = lane&15; lanes 0-15: K = kc*32+0..15,
    // lanes 16-31: K = kc*32+16..31  (contiguous, matches B-frag layout)
    bf16x16 afrag;
    if constexpr (FIRST) {
      const int b    = bbase + (lr >> 3);
      const int node = lr & 7;
      const float* pr = roi + ((size_t)(b * 8 + node)) * 512 + kc * 32 + lh * 16;
      const float4 f0 = *(const float4*)(pr);
      const float4 f1 = *(const float4*)(pr + 4);
      const float4 f2 = *(const float4*)(pr + 8);
      const float4 f3 = *(const float4*)(pr + 12);
      afrag[0]=(__bf16)f0.x;  afrag[1]=(__bf16)f0.y;  afrag[2]=(__bf16)f0.z;  afrag[3]=(__bf16)f0.w;
      afrag[4]=(__bf16)f1.x;  afrag[5]=(__bf16)f1.y;  afrag[6]=(__bf16)f1.z;  afrag[7]=(__bf16)f1.w;
      afrag[8]=(__bf16)f2.x;  afrag[9]=(__bf16)f2.y;  afrag[10]=(__bf16)f2.z; afrag[11]=(__bf16)f2.w;
      afrag[12]=(__bf16)f3.x; afrag[13]=(__bf16)f3.y; afrag[14]=(__bf16)f3.z; afrag[15]=(__bf16)f3.w;
    } else {
      const __bf16* p = xb + lr * DIN + kc * 32 + lh * 16;
      const bf16x8 lo = *(const bf16x8*)(p);
      const bf16x8 hi = *(const bf16x8*)(p + 8);
      #pragma unroll
      for (int j = 0; j < 8; ++j) { afrag[j] = lo[j]; afrag[8 + j] = hi[j]; }
    }

    const __bf16* bsrc;
    if constexpr (FIRST) {
#if USE_ASYNC
      bsrc = stage + (kc & 1) * CH;
#else
      bsrc = stage;
#endif
    } else {
      bsrc = wfrag + (size_t)kc * CH;
    }

    #pragma unroll
    for (int t = 0; t < NT; ++t) {
      const bf16x16 bfrag = *(const bf16x16*)(bsrc + ((size_t)t * 32 + lane) * 16);
      acc[t] = __builtin_amdgcn_wmma_f32_16x16x32_bf16(
          false, afrag, false, bfrag, (short)0, acc[t], false, false);
    }

    if constexpr (FIRST) __syncthreads();  // readers done before next chunk lands
  }

  // per-lane partial source/target logits; lane holds column c, rows v+8*lh
  float pls[8], plt[8], bv[NT];
  #pragma unroll
  for (int v = 0; v < 8; ++v) { pls[v] = 0.f; plt[v] = 0.f; }
  #pragma unroll
  for (int t = 0; t < NT; ++t) {
    const int c   = t * 16 + lr;
    const bool ok = (c < VD);
    const float asv = ok ? a_s[c] : 0.f;
    const float atv = ok ? a_t[c] : 0.f;
    bv[t] = ok ? bias[c] : 0.f;
    #pragma unroll
    for (int v = 0; v < 8; ++v) { pls[v] += acc[t][v] * asv; plt[v] += acc[t][v] * atv; }
  }
  #pragma unroll
  for (int v = 0; v < 8; ++v) { pls[v] = red16(pls[v]); plt[v] = red16(plt[v]); }

  float px[8], py[8], pz[8];
  if constexpr (FIRST) {
    const int b = bbase + lh;
    #pragma unroll
    for (int n = 0; n < 8; ++n) {
      const float* pp = prop + ((size_t)(b * 8 + n)) * 6;
      px[n] = (pp[0] + pp[3]) * 0.5f;
      py[n] = (pp[1] + pp[4]) * 0.5f;
      pz[n] = (pp[2] + pp[5]) * 0.5f;
    }
  }

  if constexpr (LAST) __syncthreads();   // all LDS reads done; xout aliases LDS

  // 8x8 attention per batch (lane-local: rows of this lane's half == its batch)
  #pragma unroll
  for (int r = 0; r < 8; ++r) {
    float w[8];
    float mx = -3.0e38f;
    #pragma unroll
    for (int s = 0; s < 8; ++s) {
      float lg = plt[r] + pls[s];
      if constexpr (FIRST) {
        const float dx = px[r] - px[s], dy = py[r] - py[s], dz = pz[r] - pz[s];
        lg += sqrtf(dx * dx + dy * dy + dz * dz) * edgec;
      }
      lg = (lg > 0.f) ? lg : 0.2f * lg;       // leaky relu 0.2
      if (s == r) lg = -1.0e9f;               // no self edges
      w[s] = lg;
      mx = fmaxf(mx, lg);
    }
    float sum = 0.f;
    #pragma unroll
    for (int s = 0; s < 8; ++s) { w[s] = __expf(w[s] - mx); sum += w[s]; }
    const float inv = 1.f / sum;
    #pragma unroll
    for (int s = 0; s < 8; ++s) w[s] *= inv;

    #pragma unroll
    for (int t = 0; t < NT; ++t) {
      float o = bv[t];
      #pragma unroll
      for (int s = 0; s < 8; ++s) o += w[s] * acc[t][s];
      const int c = t * 16 + lr;
      if constexpr (LAST) {
        if (c < 8)  // gnn_x: [batch][node*8 + dim] into concat slot 64..127
          xout[(lb0 + lh) * 128 + 64 + r * 8 + c] = (__bf16)o;
      } else {
        const int row = r + 8 * lh;
        xout[row * OSTR + c] = (__bf16)o;
        if constexpr (ZPAD) xout[row * OSTR + 16 + c] = (__bf16)0.f;
      }
    }
  }
}

// ---------------- main fused kernel: 8 waves x 2 batches = 16 batches/block ---
__global__ __launch_bounds__(256) void sg_main(
    const float* __restrict__ roi, const float* __restrict__ prop,
    const float* __restrict__ convx,
    const float* __restrict__ as1, const float* __restrict__ at1, const float* __restrict__ b1,
    const float* __restrict__ as2, const float* __restrict__ at2, const float* __restrict__ b2,
    const float* __restrict__ as3, const float* __restrict__ at3, const float* __restrict__ b3,
    const float* __restrict__ as4, const float* __restrict__ at4, const float* __restrict__ b4,
    const float* __restrict__ as5, const float* __restrict__ at5, const float* __restrict__ b5,
    const float* __restrict__ bfc,
    const __bf16* __restrict__ wsb, const float* __restrict__ edgep,
    float* __restrict__ out)
{
  __shared__ __align__(32) __bf16 smem[32768];   // 64 KB
  const int tid  = threadIdx.x;
  const int wave = tid >> 5;
  const int lane = tid & 31;
  const int lr = lane & 15, lh = lane >> 4;
  const int bbase = blockIdx.x * 16 + wave * 2;

  __bf16* xb    = smem + wave * 4096;            // per-wave 8 KB activation buf
  __bf16* stg   = smem;                          // layer-1 B staging (aliased:
                                                 //  used only before xb is written)
  __bf16* xfc   = smem;                          // [16][128] bf16 (aliased, post-barrier)
  float*  outfc = (float*)(smem + 4096);         // [16][64] f32
  float*  sinv  = (float*)(smem + 8192);         // [16] f32

  const float edgec = *edgep;

  gat_layer<512, 256, 256, 256, true , false, false>(nullptr, xb, wsb + OFF_W1,
      stg, tid, as1, at1, b1, roi, prop, edgec, bbase, lane, 0);
  gat_layer<256, 128, 128, 128, false, false, false>(xb, xb, wsb + OFF_W2,
      nullptr, tid, as2, at2, b2, nullptr, nullptr, 0.f, 0, lane, 0);
  gat_layer<128,  64,  64,  64, false, false, false>(xb, xb, wsb + OFF_W3,
      nullptr, tid, as3, at3, b3, nullptr, nullptr, 0.f, 0, lane, 0);
  gat_layer< 64,  16,  16,  32, false, true , false>(xb, xb, wsb + OFF_W4,
      nullptr, tid, as4, at4, b4, nullptr, nullptr, 0.f, 0, lane, 0);  // pad K for L5
  gat_layer< 32,  16,   8,   0, false, false, true >(xb, xfc, wsb + OFF_W5,
      nullptr, tid, as5, at5, b5, nullptr, nullptr, 0.f, 0, lane, wave * 2);

  // conv_x -> xfc[:, 0:64]
  #pragma unroll
  for (int i = 0; i < 4; ++i) {
    const int j = lane * 4 + i;                  // 0..127
    const int bb = j >> 6, col = j & 63;
    xfc[(wave * 2 + bb) * 128 + col] = (__bf16)convx[(size_t)(bbase + bb) * 64 + col];
  }
  __syncthreads();

  // final FC: [16,128] x [128,64] with 4 waves, one 16-col tile each
  if (wave < 4) {
    v8f acc = {};
    const __bf16* wfcf = wsb + OFF_WFC;
    for (int kc = 0; kc < 4; ++kc) {
      const __bf16* p = xfc + lr * 128 + kc * 32 + lh * 16;
      const bf16x8 lo = *(const bf16x8*)(p);
      const bf16x8 hi = *(const bf16x8*)(p + 8);
      bf16x16 afrag;
      #pragma unroll
      for (int j = 0; j < 8; ++j) { afrag[j] = lo[j]; afrag[8 + j] = hi[j]; }
      const bf16x16 bfrag =
          *(const bf16x16*)(wfcf + ((size_t)(kc * 4 + wave) * 32 + lane) * 16);
      acc = __builtin_amdgcn_wmma_f32_16x16x32_bf16(
          false, afrag, false, bfrag, (short)0, acc, false, false);
    }
    const int c = wave * 16 + lr;
    const float bvv = bfc[c];
    #pragma unroll
    for (int v = 0; v < 8; ++v) outfc[(v + 8 * lh) * 64 + c] = acc[v] + bvv;
  }
  __syncthreads();

  if (tid < 16) {
    float s = 0.f;
    #pragma unroll
    for (int c = 0; c < 64; ++c) { const float x = outfc[tid * 64 + c]; s += x * x; }
    sinv[tid] = 1.f / fmaxf(sqrtf(s), 1e-12f);
  }
  __syncthreads();

  {
    const int idx = tid * 4;
    const int row = idx >> 6, c0 = idx & 63;
    const float4 x = *(const float4*)(outfc + row * 64 + c0);
    const float iv = sinv[row];
    float4 r; r.x = x.x * iv; r.y = x.y * iv; r.z = x.z * iv; r.w = x.w * iv;
    *(float4*)(out + ((size_t)blockIdx.x * 16 + row) * 64 + c0) = r;
  }
}

// ---------------- host entry --------------------------------------------------
extern "C" void kernel_launch(void* const* d_in, const int* in_sizes, int n_in,
                              void* d_out, int out_size, void* d_ws, size_t ws_size,
                              hipStream_t stream) {
  const float* roi  = (const float*)d_in[0];
  const float* prop = (const float*)d_in[1];
  const float* convx= (const float*)d_in[2];
  const float* W1 = (const float*)d_in[3];
  const float* as1= (const float*)d_in[4];
  const float* at1= (const float*)d_in[5];
  const float* b1 = (const float*)d_in[6];
  const float* W2 = (const float*)d_in[7];
  const float* as2= (const float*)d_in[8];
  const float* at2= (const float*)d_in[9];
  const float* b2 = (const float*)d_in[10];
  const float* W3 = (const float*)d_in[11];
  const float* as3= (const float*)d_in[12];
  const float* at3= (const float*)d_in[13];
  const float* b3 = (const float*)d_in[14];
  const float* W4 = (const float*)d_in[15];
  const float* as4= (const float*)d_in[16];
  const float* at4= (const float*)d_in[17];
  const float* b4 = (const float*)d_in[18];
  const float* W5 = (const float*)d_in[19];
  const float* as5= (const float*)d_in[20];
  const float* at5= (const float*)d_in[21];
  const float* b5 = (const float*)d_in[22];
  const float* We1= (const float*)d_in[23];
  const float* ae1= (const float*)d_in[24];
  const float* Wfc= (const float*)d_in[25];
  const float* bfc= (const float*)d_in[26];

  __bf16* wsb  = (__bf16*)d_ws;
  float* edgep = (float*)((char*)d_ws + OFF_EDGE_BYTES);

  auto nb = [](int total) { return (total + 255) / 256; };
  prep_wfrag<<<nb(512 * 256), 256, 0, stream>>>(W1,  wsb + OFF_W1,  512, 256, 512, 256);
  prep_wfrag<<<nb(256 * 128), 256, 0, stream>>>(W2,  wsb + OFF_W2,  256, 128, 256, 128);
  prep_wfrag<<<nb(128 *  64), 256, 0, stream>>>(W3,  wsb + OFF_W3,  128,  64, 128,  64);
  prep_wfrag<<<nb( 64 *  16), 256, 0, stream>>>(W4,  wsb + OFF_W4,   64,  16,  64,  16);
  prep_wfrag<<<nb( 32 *  16), 256, 0, stream>>>(W5,  wsb + OFF_W5,   16,   8,  32,  16);
  prep_wfrag<<<nb(128 *  64), 256, 0, stream>>>(Wfc, wsb + OFF_WFC, 128,  64, 128,  64);
  prep_edge<<<1, 256, 0, stream>>>(We1, ae1, edgep);

  const int B = in_sizes[2] / 64;    // conv_x is [B,64]
  sg_main<<<B / 16, 256, 0, stream>>>(roi, prop, convx,
                                      as1, at1, b1, as2, at2, b2,
                                      as3, at3, b3, as4, at4, b4,
                                      as5, at5, b5, bfc, wsb, edgep,
                                      (float*)d_out);
}
